// PositionalAdjExtract_44057774522568
// MI455X (gfx1250) — matrix operand order repaired and checked by reading, not compiled, not run on hardware
//
#include <hip/hip_runtime.h>

typedef __attribute__((ext_vector_type(16))) _Float16 v16h;
typedef __attribute__((ext_vector_type(8)))  float    v8f;

#define NN     2048
#define BLK    512        // 16 waves of 32
#define JTILE  256        // columns per block-iteration (16 waves x 16 j)
#define ITERS  (NN / JTILE)

// fc[i,j] = sum_d emb[i,j,d] * W[d] + b, with emb = [sin(m_k), cos(m_k)],
// m_k = AMP[f] * pos_g,  k = g*8+f,  AMP[f] = 100 * 1000^(-f/8).
// adj[i,j] = max(fc,1e-6) / sum_j max(fc,1e-6)   (softmax(log(clip)) == normalize)
__global__ __launch_bounds__(BLK)
void PositionalAdj_wmma_kernel(const float* __restrict__ rois,
                               const float* __restrict__ Wg,
                               const float* __restrict__ Bg,
                               float* __restrict__ out)
{
    const int tid  = threadIdx.x;
    const int lane = tid & 31;
    const int wave = tid >> 5;
    const int i    = blockIdx.x;

    const bool hi = (lane & 16) != 0;   // hi lanes handle dy/dh features (K 8-15, 24-31)
    const int  p  = lane & 15;          // pair (column j) index within the wave's 16-tile

    // --- row-i box geometry (uniform across block) ---
    const float xmin_i = rois[4*i+0], ymin_i = rois[4*i+1];
    const float xmax_i = rois[4*i+2], ymax_i = rois[4*i+3];
    const float cx_i = 0.5f*(xmin_i + xmax_i);
    const float cy_i = 0.5f*(ymin_i + ymax_i);
    const float w_i  = xmax_i - xmin_i + 1.0f;
    const float h_i  = ymax_i - ymin_i + 1.0f;
    const float ci      = hi ? cy_i : cx_i;
    const float inv_i   = 1.0f / (hi ? h_i : w_i);
    const float logsz_i = __logf(hi ? h_i : w_i);

    // --- B matrices: W replicated across all 16 columns.
    // 16-bit B layout: lanes 0-15 hold K=0..7,16..23; lanes 16-31 hold K=8..15,24..31.
    v16h Bs, Bc;
    #pragma unroll
    for (int t = 0; t < 16; ++t) {
        Bs[t] = (_Float16)Wg[(hi ? 8 : 0) + (t < 8 ? t : t + 8)];        // sin dims 0..31
        Bc[t] = (_Float16)Wg[32 + (hi ? 8 : 0) + (t < 8 ? t : t + 8)];   // cos dims 32..63
    }
    const float bias = Bg[0];

    const float AMP[8] = {100.0f, 42.169650f, 17.782794f, 7.4989421f,
                          3.1622777f, 1.3335214f, 0.56234133f, 0.23713737f};

    float vals[ITERS];
    float partial = 0.0f;
    const bool primary = ((lane & 8) == 0);  // unique holder of fc[(lane&7)+8*hi]

    #pragma unroll 1
    for (int it = 0; it < ITERS; ++it) {
        const int j = it*JTILE + wave*16 + p;

        // column-j geometry: lo lanes read x_min/x_max, hi lanes y_min/y_max
        const float lo  = rois[4*j + (hi ? 1 : 0)];
        const float hg  = rois[4*j + (hi ? 3 : 2)];
        const float cj  = 0.5f*(lo + hg);
        const float szj = hg - lo + 1.0f;

        const float gA = __logf(fmaxf(fabsf((ci - cj) * inv_i), 0.001f)); // dx or dy
        const float gB = logsz_i - __logf(szj);                            // dw or dh

        // A matrices per ISA 16-bit A layout:
        //  lane<16 : elems 0-7 -> K0-7 (g=0), elems 8-15 -> K16-23 (g=2)
        //  lane>=16: elems 0-7 -> K8-15 (g=1), elems 8-15 -> K24-31 (g=3)
        v16h As, Ac;
        #pragma unroll
        for (int t = 0; t < 8; ++t) {
            const float a0 = AMP[t] * gA;
            const float a1 = AMP[t] * gB;
            As[t]     = (_Float16)__sinf(a0);
            Ac[t]     = (_Float16)__cosf(a0);
            As[t + 8] = (_Float16)__sinf(a1);
            Ac[t + 8] = (_Float16)__cosf(a1);
        }

        v8f acc = {};
        acc = __builtin_amdgcn_wmma_f32_16x16x32_f16(false, As, false, Bs,
                                                     (short)0, acc, false, false);
        acc = __builtin_amdgcn_wmma_f32_16x16x32_f16(false, Ac, false, Bc,
                                                     (short)0, acc, false, false);

        // D layout: VGPR r, lane l -> D[r + 8*(l>=16), l&15]; all columns equal.
        const int s = lane & 7;
        const float fc =
            (s & 4) ? ((s & 2) ? ((s & 1) ? acc[7] : acc[6])
                               : ((s & 1) ? acc[5] : acc[4]))
                    : ((s & 2) ? ((s & 1) ? acc[3] : acc[2])
                               : ((s & 1) ? acc[1] : acc[0]));

        const float c = fmaxf(fc + bias, 1e-6f);
        vals[it] = c;
        if (primary) partial += c;
    }

    // --- row sum across the block ---
    __shared__ float red[BLK];
    red[tid] = partial;
    __syncthreads();
    #pragma unroll
    for (int s2 = BLK/2; s2 > 0; s2 >>= 1) {
        if (tid < s2) red[tid] += red[tid + s2];
        __syncthreads();
    }
    const float rinv = 1.0f / red[0];

    // --- normalized store (primary lanes cover 16 distinct columns per wave) ---
    const int colbase = wave*16 + (lane & 7) + (hi ? 8 : 0);
    if (primary) {
        #pragma unroll
        for (int it = 0; it < ITERS; ++it) {
            out[(size_t)i * NN + it*JTILE + colbase] = vals[it] * rinv;
        }
    }
}

extern "C" void kernel_launch(void* const* d_in, const int* in_sizes, int n_in,
                              void* d_out, int out_size, void* d_ws, size_t ws_size,
                              hipStream_t stream) {
    (void)in_sizes; (void)n_in; (void)d_ws; (void)ws_size; (void)out_size;
    const float* rois = (const float*)d_in[0];
    const float* Wg   = (const float*)d_in[1];
    const float* Bg   = (const float*)d_in[2];
    float* out        = (float*)d_out;

    PositionalAdj_wmma_kernel<<<NN, BLK, 0, stream>>>(rois, Wg, Bg, out);
}
